// TransformerEncoderLayer_70282844832276
// MI455X (gfx1250) — compile-verified, matrix operand order checked by reading, not run on hardware
//
#include <hip/hip_runtime.h>
#include <cstdint>
#include <cstddef>

typedef __bf16 bf16;
typedef bf16  v16bf __attribute__((ext_vector_type(16)));
typedef bf16  v8bf  __attribute__((ext_vector_type(8)));
typedef float v8f   __attribute__((ext_vector_type(8)));

#define NTOK   20000
#define BATCH  4
#define PERB   5000
#define DMODEL 256
#define NH     8
#define DHEAD  32
#define KSA    1000
#define NLVL   4
#define NPTS   4
#define DFF    1024
#define STOT   5440
#define QPAD   1008
#define KPAD   1024
#define SCALE_ATT 0.17677669529663687f

// ---------------------------------------------------------------------------
// WMMA helpers (CDNA5 gfx1250, wave32, v_wmma_f32_16x16x32_bf16)
// A/B 16-bit fragment layout: lane&15 = row, K halves at (lane>>4)*8 and +16.
// ---------------------------------------------------------------------------
__device__ __forceinline__ v16bf load_frag(const bf16* __restrict__ base, int ld, int lane) {
    int row = lane & 15;
    int cb  = (lane >> 4) << 3;
    const bf16* p = base + (size_t)row * ld;
    v8bf lo = *(const v8bf*)(p + cb);
    v8bf hi = *(const v8bf*)(p + 16 + cb);
    v16bf f;
#pragma unroll
    for (int i = 0; i < 8; ++i) { f[i] = lo[i]; f[8 + i] = hi[i]; }
    return f;
}

// Fragment load from a per-lane row base pointer (base already includes
// laneRow*K + chunkBase); OFF is a compile-time element offset so it lowers
// into the global_load_b128 immediate offset field (no extra address VGPRs).
template<int OFF>
__device__ __forceinline__ v16bf load_frag_imm(const bf16* __restrict__ p) {
    v8bf lo = *(const v8bf*)(p + OFF);
    v8bf hi = *(const v8bf*)(p + OFF + 16);
    v16bf f;
#pragma unroll
    for (int i = 0; i < 8; ++i) { f[i] = lo[i]; f[8 + i] = hi[i]; }
    return f;
}

__device__ __forceinline__ v8f wmma_bf16(v16bf a, v16bf b, v8f c) {
    return __builtin_amdgcn_wmma_f32_16x16x32_bf16(false, a, false, b, (short)0, c, false, false);
}

__device__ __forceinline__ float gelu_tanh(float x) {
    float x3 = x * x * x;
    float t  = tanhf(0.7978845608028654f * (x + 0.044715f * x3));
    return 0.5f * x * (1.0f + t);
}

// ---------------------------------------------------------------------------
// Generic WMMA GEMM: out = act(A[M,K] * Bt[N,K]^T + bias) + resid
// One wave handles a 32x64 tile (2 A-frags x 4 B-frags = 8 WMMAs per K step).
// M multiple of 32, N multiple of 64. K is compile-time (256 or 1024) so all
// intra-tile fragment offsets are instruction immediates -> 2 address regs.
// ---------------------------------------------------------------------------
template<int K>
__global__ void gemm_kernel(const bf16* __restrict__ A, const bf16* __restrict__ Bt,
                            const float* __restrict__ bias, const float* __restrict__ resid,
                            const int* __restrict__ rowmap, float* __restrict__ outF,
                            bf16* __restrict__ outB, int M, int N, int act)
{
    int ntiles = N >> 6;
    int mtiles = M >> 5;
    int tid  = blockIdx.x * 8 + (threadIdx.x >> 5);
    if (tid >= mtiles * ntiles) return;
    int tm = tid / ntiles, tn = tid % ntiles;
    int m0 = tm << 5, n0 = tn << 6;
    int lane = threadIdx.x & 31;
    int lrow = lane & 15;
    int cb   = (lane >> 4) << 3;

    const bf16* pa = A  + (size_t)(m0 + lrow) * K + cb;
    const bf16* pb = Bt + (size_t)(n0 + lrow) * K + cb;

    v8f acc[2][4];
#pragma unroll
    for (int i = 0; i < 2; ++i)
#pragma unroll
        for (int j = 0; j < 4; ++j)
#pragma unroll
            for (int r = 0; r < 8; ++r) acc[i][j][r] = 0.0f;

    for (int kt = 0; kt < K / 32; ++kt) {
        v16bf a0 = load_frag_imm<0>(pa);
        v16bf a1 = load_frag_imm<16 * K>(pa);
        v16bf b0 = load_frag_imm<0>(pb);
        v16bf b1 = load_frag_imm<16 * K>(pb);
        v16bf b2 = load_frag_imm<32 * K>(pb);
        v16bf b3 = load_frag_imm<48 * K>(pb);
        pa += 32;
        pb += 32;
        acc[0][0] = wmma_bf16(a0, b0, acc[0][0]);
        acc[0][1] = wmma_bf16(a0, b1, acc[0][1]);
        acc[0][2] = wmma_bf16(a0, b2, acc[0][2]);
        acc[0][3] = wmma_bf16(a0, b3, acc[0][3]);
        acc[1][0] = wmma_bf16(a1, b0, acc[1][0]);
        acc[1][1] = wmma_bf16(a1, b1, acc[1][1]);
        acc[1][2] = wmma_bf16(a1, b2, acc[1][2]);
        acc[1][3] = wmma_bf16(a1, b3, acc[1][3]);
    }

    int ln = lane & 15;
    int lh = (lane >> 4) << 3;
#pragma unroll
    for (int i = 0; i < 2; ++i)
#pragma unroll
        for (int j = 0; j < 4; ++j)
#pragma unroll
            for (int r = 0; r < 8; ++r) {
                int m = m0 + i * 16 + lh + r;
                int n = n0 + j * 16 + ln;
                float v = acc[i][j][r];
                if (bias)  v += bias[n];
                if (act == 1) v = gelu_tanh(v);
                if (resid) v += resid[(size_t)m * N + n];
                int row = rowmap ? rowmap[m] : m;
                if (outF) outF[(size_t)row * N + n] = v;
                else      outB[(size_t)row * N + n] = (bf16)v;
            }
}

// ---------------------------------------------------------------------------
// Deterministic per-batch radix-select top-K (K=1000 of 5000), ordered output.
// ---------------------------------------------------------------------------
__global__ void topk_kernel(const float* __restrict__ esc, const float* __restrict__ psal,
                            const int* __restrict__ qbo, unsigned* __restrict__ keys,
                            int* __restrict__ tk)
{
    int b = blockIdx.x;
    int t = threadIdx.x;
    int start = qbo[b], cnt = qbo[b + 1] - qbo[b];
    __shared__ unsigned hist[256];
    __shared__ int wsum[8];
    __shared__ unsigned s_prefix;
    __shared__ int s_need, s_base;

    for (int i = t; i < cnt; i += 256) {
        float s = esc[start + i] + psal[start + i];
        unsigned u = __float_as_uint(s);
        u = (u & 0x80000000u) ? ~u : (u | 0x80000000u);
        keys[start + i] = u;
    }
    if (t == 0) { s_prefix = 0u; s_need = KSA; s_base = 0; }
    __syncthreads();

    for (int p = 0; p < 4; ++p) {
        hist[t] = 0u;
        __syncthreads();
        int shift = 24 - 8 * p;
        unsigned pref = s_prefix;
        for (int i = t; i < cnt; i += 256) {
            unsigned key = keys[start + i];
            bool ok = (p == 0) || ((key >> (shift + 8)) == pref);
            if (ok) atomicAdd(&hist[(key >> shift) & 255u], 1u);
        }
        __syncthreads();
        if (t == 0) {
            unsigned cum = 0; int need = s_need; unsigned chosen = 0;
            for (int v = 255; v >= 0; --v) {
                unsigned c = hist[v];
                if (cum + c >= (unsigned)need) { chosen = (unsigned)v; need -= (int)cum; break; }
                cum += c;
            }
            s_prefix = (pref << 8) | chosen;
            s_need = need;
        }
        __syncthreads();
    }
    unsigned T = s_prefix;
    __syncthreads();

    int lane = t & 31, wid = t >> 5;
    int nchunks = (cnt + 255) / 256;
    // phase 1: strictly greater, phase 2: equal (take first s_need)
    for (int ph = 0; ph < 2; ++ph) {
        for (int c = 0; c < nchunks; ++c) {
            int i = c * 256 + t;
            bool flag;
            if (i < cnt) {
                unsigned key = keys[start + i];
                flag = ph == 0 ? (key > T) : (key == T);
            } else flag = false;
            unsigned bal = (unsigned)__ballot(flag);
            int lcnt = __popc(bal & ((1u << lane) - 1u));
            if (lane == 0) wsum[wid] = __popc(bal);
            __syncthreads();
            int woff = 0, tot = 0;
            for (int w = 0; w < 8; ++w) { if (w < wid) woff += wsum[w]; tot += wsum[w]; }
            int pos = s_base + woff + lcnt;
            if (flag && pos < KSA) tk[b * KSA + pos] = start + i;
            __syncthreads();
            if (t == 0) s_base += tot;
            __syncthreads();
        }
    }
}

// ---------------------------------------------------------------------------
// LayerNorm (256 threads = one row). Reductions via wave32 shfl + shared.
// ---------------------------------------------------------------------------
__device__ __forceinline__ float block_sum256(float v, float* red) {
#pragma unroll
    for (int o = 16; o > 0; o >>= 1) v += __shfl_xor(v, o, 32);
    int lane = threadIdx.x & 31, wid = threadIdx.x >> 5;
    if (lane == 0) red[wid] = v;
    __syncthreads();
    float s = 0.f;
#pragma unroll
    for (int w = 0; w < 8; ++w) s += red[w];
    __syncthreads();
    return s;
}

__global__ void ln_rows_kernel(const float* __restrict__ in, const float* __restrict__ g,
                               const float* __restrict__ bb, bf16* __restrict__ out)
{
    __shared__ float red[8];
    int row = blockIdx.x, t = threadIdx.x;
    float v = in[(size_t)row * DMODEL + t];
    float mu = block_sum256(v, red) * (1.0f / DMODEL);
    float dv = v - mu;
    float var = block_sum256(dv * dv, red) * (1.0f / DMODEL);
    float rs = rsqrtf(var + 1e-5f);
    out[(size_t)row * DMODEL + t] = (bf16)(dv * rs * g[t] + bb[t]);
}

__global__ void ln_gather_kernel(const float* __restrict__ queries, const int* __restrict__ tk,
                                 const float* __restrict__ g, const float* __restrict__ bb,
                                 float* __restrict__ xg, bf16* __restrict__ out)
{
    __shared__ float red[8];
    int row = blockIdx.x, t = threadIdx.x;
    int tok = tk[row];
    float v = queries[(size_t)tok * DMODEL + t];
    xg[(size_t)row * DMODEL + t] = v;
    float mu = block_sum256(v, red) * (1.0f / DMODEL);
    float dv = v - mu;
    float var = block_sum256(dv * dv, red) * (1.0f / DMODEL);
    float rs = rsqrtf(var + 1e-5f);
    out[(size_t)row * DMODEL + t] = (bf16)(dv * rs * g[t] + bb[t]);
}

// ---------------------------------------------------------------------------
// RoPE + repack q,k row-major bf16 (padded), v transposed (DH x KPAD).
// blockIdx = b*1024 + i ; threads: h = t>>5, d = t&31.
// ---------------------------------------------------------------------------
__global__ void rope_prep_kernel(const float* __restrict__ qkv, const int* __restrict__ tk,
                                 const int* __restrict__ qsi,
                                 bf16* __restrict__ Qr, bf16* __restrict__ Kr, bf16* __restrict__ Vt)
{
    int bi = blockIdx.x;
    int b = bi >> 10, i = bi & 1023;
    int t = threadIdx.x;
    int h = t >> 5, d = t & 31;
    int bh = b * NH + h;
    if (i >= KSA) {
        if (i < QPAD) Qr[((size_t)bh * QPAD + i) * DHEAD + d] = (bf16)0.f;
        Kr[((size_t)bh * KPAD + i) * DHEAD + d] = (bf16)0.f;
        Vt[((size_t)bh * DHEAD + d) * KPAD + i] = (bf16)0.f;
        return;
    }
    int row = b * KSA + i;
    int tok = tk[row];
    float py = (float)qsi[1 * NTOK + tok];
    float px = (float)qsi[2 * NTOK + tok];
    float lv = (float)qsi[3 * NTOK + tok];
    int pi = d >> 1;
    float ang;
    if (pi < 6)       ang = py * __powf(100.0f, -(float)pi / 6.0f);
    else if (pi < 12) ang = px * __powf(100.0f, -(float)(pi - 6) / 6.0f);
    else              ang = lv * __powf(10.0f,  -(float)(pi - 12) / 4.0f);
    float cs = cosf(ang), sn = sinf(ang);
    const float* base = qkv + (size_t)row * (3 * DMODEL) + h * DHEAD + (pi << 1);
    float q0 = base[0],       q1 = base[1];
    float k0 = base[DMODEL],  k1 = base[DMODEL + 1];
    float qv = (d & 1) ? (q0 * sn + q1 * cs) : (q0 * cs - q1 * sn);
    float kv = (d & 1) ? (k0 * sn + k1 * cs) : (k0 * cs - k1 * sn);
    float vv = qkv[(size_t)row * (3 * DMODEL) + 2 * DMODEL + h * DHEAD + d];
    Qr[((size_t)bh * QPAD + i) * DHEAD + d] = (bf16)qv;
    Kr[((size_t)bh * KPAD + i) * DHEAD + d] = (bf16)kv;
    Vt[((size_t)bh * DHEAD + d) * KPAD + i] = (bf16)vv;
}

// ---------------------------------------------------------------------------
// Flash-style attention, all WMMA. One wave = one 16-query tile of one (b,h).
// S^T = Kfrag x Qfrag ; O^T += V^Tfrag x P^Tfrag with online softmax.
// ---------------------------------------------------------------------------
#define QTILES 63  // QPAD/16
__global__ void attention_kernel(const bf16* __restrict__ Qr, const bf16* __restrict__ Kr,
                                 const bf16* __restrict__ Vt, float* __restrict__ o)
{
    int gw = blockIdx.x * 8 + (threadIdx.x >> 5);
    if (gw >= BATCH * NH * QTILES) return;
    int bh = gw / QTILES;
    int qt = gw % QTILES;
    int lane = threadIdx.x & 31;
    int lh = (lane >> 4) << 3;

    v16bf qf = load_frag(Qr + ((size_t)bh * QPAD + qt * 16) * DHEAD, DHEAD, lane);

    v8f ot0, ot1, zf;
#pragma unroll
    for (int r = 0; r < 8; ++r) { ot0[r] = 0.f; ot1[r] = 0.f; zf[r] = 0.f; }
    float m = -1e30f, lsum = 0.f;

    for (int kt = 0; kt < KPAD / 32; ++kt) {
        int k0 = kt * 32;
        const bf16* kb = Kr + ((size_t)bh * KPAD + k0) * DHEAD;
        v16bf ka0 = load_frag(kb, DHEAD, lane);
        v16bf ka1 = load_frag(kb + 16 * DHEAD, DHEAD, lane);
        v8f st0 = wmma_bf16(ka0, qf, zf);
        v8f st1 = wmma_bf16(ka1, qf, zf);

        float s0[8], s1[8];
        float mloc = -1e30f;
#pragma unroll
        for (int r = 0; r < 8; ++r) {
            int ka = k0 + lh + r;
            int kbb = k0 + 16 + lh + r;
            s0[r] = (ka  < KSA) ? st0[r] * SCALE_ATT : -1e30f;
            s1[r] = (kbb < KSA) ? st1[r] * SCALE_ATT : -1e30f;
            mloc = fmaxf(mloc, fmaxf(s0[r], s1[r]));
        }
        mloc = fmaxf(mloc, __shfl_xor(mloc, 16, 32));
        float mnew = fmaxf(m, mloc);
        float alpha = __expf(m - mnew);
        float p0[8], p1[8], psum = 0.f;
#pragma unroll
        for (int r = 0; r < 8; ++r) {
            p0[r] = __expf(s0[r] - mnew);
            p1[r] = __expf(s1[r] - mnew);
            psum += p0[r] + p1[r];
        }
        psum += __shfl_xor(psum, 16, 32);
        lsum = lsum * alpha + psum;
        m = mnew;
#pragma unroll
        for (int r = 0; r < 8; ++r) { ot0[r] *= alpha; ot1[r] *= alpha; }

        v16bf pb;
#pragma unroll
        for (int r = 0; r < 8; ++r) { pb[r] = (bf16)p0[r]; pb[8 + r] = (bf16)p1[r]; }

        const bf16* vb = Vt + (size_t)bh * DHEAD * KPAD + k0;
        v16bf va0 = load_frag(vb, KPAD, lane);
        v16bf va1 = load_frag(vb + (size_t)16 * KPAD, KPAD, lane);
        ot0 = wmma_bf16(va0, pb, ot0);
        ot1 = wmma_bf16(va1, pb, ot1);
    }

    float inv = 1.0f / lsum;
    int q = qt * 16 + (lane & 15);
    if (q < KSA) {
        int b = bh >> 3, h = bh & 7;
        float* ob = o + (((size_t)b * KSA + q) * NH + h) * DHEAD;
#pragma unroll
        for (int r = 0; r < 8; ++r) {
            ob[lh + r]      = ot0[r] * inv;
            ob[16 + lh + r] = ot1[r] * inv;
        }
    }
}

// ---------------------------------------------------------------------------
// Per-row softmax over 16 samples per head (aw).
// ---------------------------------------------------------------------------
__global__ void aw_softmax_kernel(float* __restrict__ aw)
{
    int n = blockIdx.x, t = threadIdx.x; // 128 threads: h = t>>4
    float v = aw[(size_t)n * 128 + t];
    float mx = v;
#pragma unroll
    for (int o = 8; o > 0; o >>= 1) mx = fmaxf(mx, __shfl_xor(mx, o, 16));
    float e = __expf(v - mx);
    float s = e;
#pragma unroll
    for (int o = 8; o > 0; o >>= 1) s += __shfl_xor(s, o, 16);
    aw[(size_t)n * 128 + t] = e / s;
}

// ---------------------------------------------------------------------------
// Multi-scale deformable sampling. One wave = one (n,h); lane = dh channel.
// ---------------------------------------------------------------------------
__global__ void msdeform_kernel(const float* __restrict__ off, const float* __restrict__ aw,
                                const bf16* __restrict__ val, const int* __restrict__ qsi,
                                const int* __restrict__ shapes, const int* __restrict__ qbo,
                                bf16* __restrict__ msd)
{
    int g = blockIdx.x * 8 + (threadIdx.x >> 5);
    if (g >= NTOK * NH) return;
    int n = g >> 3, h = g & 7;
    int d = threadIdx.x & 31;

    int b = 0;
#pragma unroll
    for (int i = 1; i < BATCH; ++i) if (n >= qbo[i]) b = i;

    int iy  = qsi[1 * NTOK + n];
    int ix  = qsi[2 * NTOK + n];
    int qlv = qsi[3 * NTOK + n];
    float refy = ((float)iy + 0.5f) / (float)shapes[qlv * 2 + 0];
    float refx = ((float)ix + 0.5f) / (float)shapes[qlv * 2 + 1];

    float acc = 0.f;
    int lev0 = 0;
    for (int l = 0; l < NLVL; ++l) {
        int Hl = shapes[l * 2], Wl = shapes[l * 2 + 1];
        float fH = (float)Hl, fW = (float)Wl;
        for (int p = 0; p < NPTS; ++p) {
            size_t oidx = (((size_t)n * NH + h) * 16 + l * NPTS + p) * 2;
            float oy = off[oidx], ox = off[oidx + 1];
            float piy = (refy + oy / fH) * fH - 0.5f;
            float pix = (refx + ox / fW) * fW - 0.5f;
            float y0 = floorf(piy), x0 = floorf(pix);
            float wy = piy - y0, wx = pix - x0;
            float a = aw[((size_t)n * NH + h) * 16 + l * NPTS + p];
#pragma unroll
            for (int c = 0; c < 4; ++c) {
                int dy = c >> 1, dx = c & 1;
                float yy = y0 + (float)dy, xx = x0 + (float)dx;
                if (yy >= 0.f && yy < fH && xx >= 0.f && xx < fW) {
                    int yi = (int)yy, xi = (int)xx;
                    float w = (dy ? wy : 1.f - wy) * (dx ? wx : 1.f - wx);
                    int flat = lev0 + yi * Wl + xi;
                    float gv = (float)val[(((size_t)b * STOT + flat) * NH + h) * DHEAD + d];
                    acc += a * w * gv;
                }
            }
        }
        lev0 += Hl * Wl;
    }
    msd[((size_t)n * NH + h) * DHEAD + d] = (bf16)acc;
}

// ---------------------------------------------------------------------------
// Utility converts
// ---------------------------------------------------------------------------
__global__ void transpose_bf16_kernel(const float* __restrict__ w, bf16* __restrict__ wt,
                                      int K, int N)
{
    size_t i = (size_t)blockIdx.x * 256 + threadIdx.x;
    if (i < (size_t)K * N) {
        int k = (int)(i / N), n = (int)(i % N);
        wt[(size_t)n * K + k] = (bf16)w[i];
    }
}

__global__ void cvt_bf16_kernel(const float* __restrict__ in, bf16* __restrict__ out, size_t n)
{
    size_t i = (size_t)blockIdx.x * 256 + threadIdx.x;
    if (i < n) out[i] = (bf16)in[i];
}

// ---------------------------------------------------------------------------
// Host launcher
// ---------------------------------------------------------------------------
extern "C" void kernel_launch(void* const* d_in, const int* in_sizes, int n_in,
                              void* d_out, int out_size, void* d_ws, size_t ws_size,
                              hipStream_t stream)
{
    (void)in_sizes; (void)n_in; (void)out_size; (void)ws_size;
    const float* queries = (const float*)d_in[0];
    const int*   qbo     = (const int*)  d_in[1];
    const float* psal    = (const float*)d_in[2];
    const int*   qsi     = (const int*)  d_in[3];
    const float* sfm     = (const float*)d_in[4];
    const int*   shapes  = (const int*)  d_in[5];
    const float* esc     = (const float*)d_in[6];
    const float* w_qkv   = (const float*)d_in[7];
    const float* w_out   = (const float*)d_in[8];
    const float* ln1g    = (const float*)d_in[9];
    const float* ln1b    = (const float*)d_in[10];
    const float* w_value = (const float*)d_in[11];
    const float* w_off   = (const float*)d_in[12];
    const float* b_off   = (const float*)d_in[13];
    const float* w_aw    = (const float*)d_in[14];
    const float* b_aw    = (const float*)d_in[15];
    const float* w_out2  = (const float*)d_in[16];
    const float* ln2g    = (const float*)d_in[17];
    const float* ln2b    = (const float*)d_in[18];
    const float* w1      = (const float*)d_in[19];
    const float* b1      = (const float*)d_in[20];
    const float* w2      = (const float*)d_in[21];
    const float* b2      = (const float*)d_in[22];
    const float* ln3g    = (const float*)d_in[23];
    const float* ln3b    = (const float*)d_in[24];

    float* out = (float*)d_out;  // working queries buffer (N x D)

    char* ws = (char*)d_ws;
    size_t cur = 0;
    auto alloc = [&](size_t bytes) -> void* {
        cur = (cur + 255) & ~(size_t)255;
        void* p = ws + cur;
        cur += bytes;
        return p;
    };

    // persistent
    int*      tk    = (int*)     alloc((size_t)BATCH * KSA * 4);
    unsigned* keys  = (unsigned*)alloc((size_t)NTOK * 4);
    bf16* wqkvT = (bf16*)alloc((size_t)768  * 256 * 2);
    bf16* woutT = (bf16*)alloc((size_t)256  * 256 * 2);
    bf16* wvalT = (bf16*)alloc((size_t)256  * 256 * 2);
    bf16* woffT = (bf16*)alloc((size_t)256  * 256 * 2);
    bf16* wawT  = (bf16*)alloc((size_t)128  * 256 * 2);
    bf16* wo2T  = (bf16*)alloc((size_t)256  * 256 * 2);
    bf16* w1T   = (bf16*)alloc((size_t)1024 * 256 * 2);
    bf16* w2T   = (bf16*)alloc((size_t)256  * 1024 * 2);
    size_t ubase = cur;

    // phase A (top-k attention)
    cur = ubase;
    float* xg   = (float*)alloc((size_t)BATCH * KSA * DMODEL * 4);
    bf16*  xn1  = (bf16*) alloc((size_t)BATCH * KSA * DMODEL * 2);
    float* qkvb = (float*)alloc((size_t)BATCH * KSA * 3 * DMODEL * 4);
    bf16*  Qr   = (bf16*) alloc((size_t)BATCH * NH * QPAD * DHEAD * 2);
    bf16*  Kr   = (bf16*) alloc((size_t)BATCH * NH * KPAD * DHEAD * 2);
    bf16*  Vt   = (bf16*) alloc((size_t)BATCH * NH * DHEAD * KPAD * 2);
    float* ob   = (float*)alloc((size_t)BATCH * KSA * DMODEL * 4);
    bf16*  obf  = (bf16*) alloc((size_t)BATCH * KSA * DMODEL * 2);

    // phase B (deformable) — aliases phase A region (time-disjoint)
    cur = ubase;
    bf16*  xn2  = (bf16*) alloc((size_t)NTOK * DMODEL * 2);
    float* offb = (float*)alloc((size_t)NTOK * 256 * 4);
    float* awb  = (float*)alloc((size_t)NTOK * 128 * 4);
    bf16*  sfmb = (bf16*) alloc((size_t)BATCH * STOT * DMODEL * 2);
    bf16*  valb = (bf16*) alloc((size_t)BATCH * STOT * DMODEL * 2);
    bf16*  msd  = (bf16*) alloc((size_t)NTOK * DMODEL * 2);

    // phase C (FFN) — aliases again
    cur = ubase;
    bf16* xn3 = (bf16*)alloc((size_t)NTOK * DMODEL * 2);
    bf16* hid = (bf16*)alloc((size_t)NTOK * DFF * 2);

    auto launch_gemm = [&](const bf16* A, const bf16* Bt, const float* bias,
                           const float* resid, const int* rowmap, float* oF, bf16* oB,
                           int M, int N, int K, int act) {
        int tiles = (M / 32) * (N / 64);
        int blocks = (tiles + 7) / 8;
        if (K == 256)
            gemm_kernel<256><<<blocks, 256, 0, stream>>>(A, Bt, bias, resid, rowmap, oF, oB, M, N, act);
        else
            gemm_kernel<1024><<<blocks, 256, 0, stream>>>(A, Bt, bias, resid, rowmap, oF, oB, M, N, act);
    };

    // 0) init working queries
    hipMemcpyAsync(out, queries, (size_t)NTOK * DMODEL * 4, hipMemcpyDeviceToDevice, stream);

    // 1) weights -> transposed bf16
    {
        struct { const float* w; bf16* wt; int K, N; } wl[8] = {
            {w_qkv, wqkvT, 256, 768}, {w_out, woutT, 256, 256},
            {w_value, wvalT, 256, 256}, {w_off, woffT, 256, 256},
            {w_aw, wawT, 256, 128}, {w_out2, wo2T, 256, 256},
            {w1, w1T, 256, 1024}, {w2, w2T, 1024, 256}};
        for (int i = 0; i < 8; ++i) {
            size_t n = (size_t)wl[i].K * wl[i].N;
            transpose_bf16_kernel<<<(unsigned)((n + 255) / 256), 256, 0, stream>>>(
                wl[i].w, wl[i].wt, wl[i].K, wl[i].N);
        }
    }

    // 2) top-k per batch
    topk_kernel<<<BATCH, 256, 0, stream>>>(esc, psal, qbo, keys, tk);

    // 3) gather + LN1
    ln_gather_kernel<<<BATCH * KSA, 256, 0, stream>>>(queries, tk, ln1g, ln1b, xg, xn1);

    // 4) qkv projection
    launch_gemm(xn1, wqkvT, nullptr, nullptr, nullptr, qkvb, nullptr, BATCH * KSA, 768, 256, 0);

    // 5) RoPE + repack
    rope_prep_kernel<<<BATCH * 1024, 256, 0, stream>>>(qkvb, tk, qsi, Qr, Kr, Vt);

    // 6) attention (flash, WMMA)
    {
        int waves = BATCH * NH * QTILES;
        attention_kernel<<<(waves + 7) / 8, 256, 0, stream>>>(Qr, Kr, Vt, ob);
    }

    // 7) o -> bf16, out projection + residual x, scatter into working queries
    {
        size_t n = (size_t)BATCH * KSA * DMODEL;
        cvt_bf16_kernel<<<(unsigned)((n + 255) / 256), 256, 0, stream>>>(ob, obf, n);
    }
    launch_gemm(obf, woutT, nullptr, xg, tk, out, nullptr, BATCH * KSA, 256, 256, 0);

    // 8) LN2 over all queries
    ln_rows_kernel<<<NTOK, 256, 0, stream>>>(out, ln2g, ln2b, xn2);

    // 9) offset & attention-weight projections
    launch_gemm(xn2, woffT, b_off, nullptr, nullptr, offb, nullptr, NTOK, 256, 256, 0);
    launch_gemm(xn2, wawT, b_aw, nullptr, nullptr, awb, nullptr, NTOK, 128, 256, 0);
    aw_softmax_kernel<<<NTOK, 128, 0, stream>>>(awb);

    // 10) value projection (feature maps)
    {
        size_t n = (size_t)BATCH * STOT * DMODEL;
        cvt_bf16_kernel<<<(unsigned)((n + 255) / 256), 256, 0, stream>>>(sfm, sfmb, n);
    }
    launch_gemm(sfmb, wvalT, nullptr, nullptr, nullptr, nullptr, valb, BATCH * STOT, 256, 256, 0);

    // 11) deformable sampling
    msdeform_kernel<<<(NTOK * NH) / 8, 256, 0, stream>>>(offb, awb, valb, qsi, shapes, qbo, msd);

    // 12) output projection + residual
    launch_gemm(msd, wo2T, nullptr, out, nullptr, out, nullptr, NTOK, 256, 256, 0);

    // 13) FFN
    ln_rows_kernel<<<NTOK, 256, 0, stream>>>(out, ln3g, ln3b, xn3);
    launch_gemm(xn3, w1T, b1, nullptr, nullptr, nullptr, hid, NTOK, 1024, 256, 1);
    launch_gemm(hid, w2T, b2, out, nullptr, out, nullptr, NTOK, 256, 1024, 0);
}
